// TrafficModel_25280177504784
// MI455X (gfx1250) — matrix operand order repaired and checked by reading, not compiled, not run on hardware
//
#include <hip/hip_runtime.h>
#include <hip/hip_bf16.h>
#include <math.h>

// ---------------- model dims ----------------
constexpr int kB   = 2;
constexpr int kT   = 256;
constexpr int kN   = 1024;
constexpr int kS   = kB * kN;      // 2048 sequences / (batch,node) pairs
constexpr int kFB  = 129;          // rfft bins (T/2+1)
constexpr int kFH  = 64;           // encoder width
constexpr int kFF  = 256;          // encoder FF width
constexpr int kNH  = 4;            // heads
constexpr int kHD  = 16;           // head dim
constexpr int kGIN = 32;
constexpr int kLIN = 16;

constexpr int kSEQ  = 129;
constexpr int kSEQP = 144;         // 9 WMMA M-tiles
constexpr int kKATT = 160;         // key length padded to 5 K-steps of 32

typedef _Float16 h16v __attribute__((ext_vector_type(16)));
typedef float    f8v  __attribute__((ext_vector_type(8)));

__device__ __forceinline__ f8v wmma16(h16v a, h16v b, f8v c) {
  // v_wmma_f32_16x16x32_f16  D = A(16x32) * B(32x16) + C
  return __builtin_amdgcn_wmma_f32_16x16x32_f16(false, a, false, b, (short)0, c, false, false);
}

// A fragment: 16x32 f16 tile, rows row0..row0+15 of row-major [*, ld] LDS array.
// Contiguous k order (matches B loaders): lanes 0-15 = M rows (K 0..15), lanes 16-31 K 16..31.
__device__ __forceinline__ h16v ldsA(const _Float16* p, int row0, int ld, int k0) {
  const int lane = threadIdx.x & 31;
  const int m  = row0 + (lane & 15);
  const int hi = lane >> 4;
  h16v a;
#pragma unroll
  for (int v = 0; v < 8; ++v) {
    int k = k0 + 2 * v + hi * 16;
    a[2 * v]     = p[m * ld + k];
    a[2 * v + 1] = p[m * ld + k + 1];
  }
  return a;
}

// B fragment (32x16) where matmul B[k][n] = W[n][k]; W row-major [N][K] (weight layout).
// Lanes 0-15 carry n (k 0..15), lanes 16-31 carry k 16..31; k-pairs contiguous in W rows.
__device__ __forceinline__ h16v ldsBw(const _Float16* p, int n0, int ld, int k0) {
  const int lane = threadIdx.x & 31;
  const int n  = n0 + (lane & 15);
  const int hi = lane >> 4;
  h16v b;
#pragma unroll
  for (int v = 0; v < 8; ++v) {
    int k = k0 + 2 * v + hi * 16;
    b[2 * v]     = p[n * ld + k];
    b[2 * v + 1] = p[n * ld + k + 1];
  }
  return b;
}

// B fragment from [K][N] row-major storage (B[k][n] = p[k*ld+n]).
__device__ __forceinline__ h16v ldsBkn(const _Float16* p, int n0, int ld, int k0) {
  const int lane = threadIdx.x & 31;
  const int n  = n0 + (lane & 15);
  const int hi = lane >> 4;
  h16v b;
#pragma unroll
  for (int v = 0; v < 8; ++v) {
    int k = k0 + 2 * v + hi * 16;
    b[2 * v]     = p[k * ld + n];
    b[2 * v + 1] = p[(k + 1) * ld + n];
  }
  return b;
}

__device__ __forceinline__ float sigmoidf(float x) { return 1.f / (1.f + __expf(-x)); }

// ---------------- 1) direct rDFT: magnitude + phase ----------------
__global__ void __launch_bounds__(256) dft_kernel(const float* __restrict__ x,
                                                  float* __restrict__ magF,
                                                  float* __restrict__ phF) {
  const int s = blockIdx.x;            // b*N + n
  const int b = s / kN, n = s % kN;
  __shared__ float xs[kT];
  for (int t = threadIdx.x; t < kT; t += 256) xs[t] = x[((size_t)b * kT + t) * kN + n];
  __syncthreads();
  for (int f = threadIdx.x; f < kFB; f += 256) {
    float re = 0.f, im = 0.f;
    const float w = -2.0f * 3.14159265358979323846f * (float)f / (float)kT;
    for (int t = 0; t < kT; ++t) {
      float sv, cv;
      __sincosf(w * (float)t, &sv, &cv);
      re += xs[t] * cv;
      im += xs[t] * sv;
    }
    magF[s * kFB + f] = sqrtf(re * re + im * im);
    phF[s * kFB + f]  = atan2f(im, re);
  }
}

// ---------------- 2) transformer encoder (whole sequence in LDS) ----------------
struct EncLayerP {
  const float *qkv_w, *qkv_b, *out_w, *out_b;
  const float *ln1_g, *ln1_b, *ln2_g, *ln2_b;
  const float *ff1_w, *ff1_b, *ff2_w, *ff2_b;
};
struct EncP {
  const float *in_w, *in_b, *pos;
  EncLayerP L[2];
  const float *outp_w, *outp_b;
};

constexpr int ENC_TOK = 0;                            // f32 [144][64]
constexpr int ENC_H   = ENC_TOK + kSEQP * kFH * 4;    // f16 [144][64]
constexpr int ENC_W   = ENC_H + kSEQP * kFH * 2;      // f16 weights (<=16384 elems)
constexpr int ENC_QKV = ENC_W + 16384 * 2;            // f16 [144][192]
constexpr int ENC_SCR = ENC_QKV + kSEQP * 192 * 2;    // f32 scores[144][144] | f16 ffh[144][256]
constexpr int ENC_ATT = ENC_SCR + kSEQP * kSEQP * 4;  // f16 att[144][160] (also Q/K staging)
constexpr int ENC_LDS = ENC_ATT + kSEQP * kKATT * 2;  // = 272384 bytes

__device__ void enc_layernorm(const float* __restrict__ tok, _Float16* __restrict__ hb,
                              const float* __restrict__ g, const float* __restrict__ bb) {
  const int tid = threadIdx.x;
  if (tid < kSEQP) {
    const int f = tid;
    if (f < kSEQ) {
      float m = 0.f;
      for (int c = 0; c < kFH; ++c) m += tok[f * kFH + c];
      m *= (1.f / kFH);
      float v = 0.f;
      for (int c = 0; c < kFH; ++c) { float d = tok[f * kFH + c] - m; v += d * d; }
      const float r = rsqrtf(v * (1.f / kFH) + 1e-5f);
      for (int c = 0; c < kFH; ++c)
        hb[f * kFH + c] = (_Float16)((tok[f * kFH + c] - m) * r * g[c] + bb[c]);
    } else {
      for (int c = 0; c < kFH; ++c) hb[f * kFH + c] = (_Float16)0.f;  // zero pad rows
    }
  }
}

__global__ void __launch_bounds__(256) encoder_kernel(const float* __restrict__ feat,
                                                      EncP p, float* __restrict__ out) {
  extern __shared__ char smem[];
  float*    tok = (float*)(smem + ENC_TOK);
  _Float16* hb  = (_Float16*)(smem + ENC_H);
  _Float16* wb  = (_Float16*)(smem + ENC_W);
  _Float16* qkv = (_Float16*)(smem + ENC_QKV);
  float*    scr = (float*)(smem + ENC_SCR);
  _Float16* ffh = (_Float16*)(smem + ENC_SCR);
  _Float16* att = (_Float16*)(smem + ENC_ATT);
  _Float16* Qs  = att;                 // [144][32] staging (overwritten by att after softmax)
  _Float16* Ks  = att + kSEQP * 32;    // [144][32]
  _Float16* Vs  = wb;                  // [160][16] staging in (free) weight buffer

  const int tid = threadIdx.x, wave = tid >> 5, lane = tid & 31;
  const int s = blockIdx.x;

  // token init: feat*in_w + in_b + pos (pad rows zero)
  for (int i = tid; i < kSEQP * kFH; i += 256) {
    int f = i / kFH, c = i % kFH;
    tok[i] = (f < kSEQ) ? feat[s * kFB + f] * p.in_w[c] + p.in_b[c] + p.pos[f * kFH + c] : 0.f;
  }
  __syncthreads();

  for (int l = 0; l < 2; ++l) {
    EncLayerP lp = p.L[l];
    // ---- LN1 -> hb; load qkv_w ----
    enc_layernorm(tok, hb, lp.ln1_g, lp.ln1_b);
    for (int i = tid; i < 192 * kFH; i += 256) wb[i] = (_Float16)lp.qkv_w[i];
    __syncthreads();

    // ---- qkv = hb @ qkv_w^T + qkv_b  (9x12 tiles, K=64) ----
    for (int tix = wave; tix < 9 * 12; tix += 8) {
      int mi = tix / 12, nj = tix % 12;
      f8v acc = {};
      for (int k0 = 0; k0 < kFH; k0 += 32)
        acc = wmma16(ldsA(hb, mi * 16, kFH, k0), ldsBw(wb, nj * 16, kFH, k0), acc);
      int nn = nj * 16 + (lane & 15), hi = lane >> 4;
      float bias = lp.qkv_b[nn];
#pragma unroll
      for (int v = 0; v < 8; ++v)
        qkv[(mi * 16 + v + hi * 8) * 192 + nn] = (_Float16)(acc[v] + bias);
    }
    __syncthreads();

    // ---- attention, one head at a time ----
    for (int h = 0; h < kNH; ++h) {
      // stage Q,K (K-dim padded 16->32 with zeros) and V (rows padded to 160)
      for (int i = tid; i < kSEQP * 32; i += 256) {
        int f = i >> 5, d = i & 31;
        Qs[i] = (d < kHD) ? qkv[f * 192 + h * kHD + d] : (_Float16)0.f;
        Ks[i] = (d < kHD) ? qkv[f * 192 + kFH + h * kHD + d] : (_Float16)0.f;
      }
      for (int i = tid; i < kKATT * kHD; i += 256) {
        int kk = i >> 4, d = i & 15;
        Vs[i] = (kk < kSEQ) ? qkv[kk * 192 + 2 * kFH + h * kHD + d] : (_Float16)0.f;
      }
      __syncthreads();
      // S = Q K^T / 4 (9x9 tiles, single K-step)
      for (int tix = wave; tix < 81; tix += 8) {
        int mi = tix / 9, nj = tix % 9;
        f8v acc = {};
        acc = wmma16(ldsA(Qs, mi * 16, 32, 0), ldsBw(Ks, nj * 16, 32, 0), acc);
        int nn = nj * 16 + (lane & 15), hi = lane >> 4;
#pragma unroll
        for (int v = 0; v < 8; ++v)
          scr[(mi * 16 + v + hi * 8) * kSEQP + nn] = acc[v] * 0.25f;
      }
      __syncthreads();
      // row softmax (f32) -> att f16, zero padded cols/rows
      if (tid < kSEQP) {
        int i = tid;
        if (i < kSEQ) {
          float mx = -3.0e38f;
          for (int j = 0; j < kSEQ; ++j) mx = fmaxf(mx, scr[i * kSEQP + j]);
          float sum = 0.f;
          for (int j = 0; j < kSEQ; ++j) {
            float e = __expf(scr[i * kSEQP + j] - mx);
            scr[i * kSEQP + j] = e;
            sum += e;
          }
          float inv = 1.f / sum;
          for (int j = 0; j < kKATT; ++j)
            att[i * kKATT + j] = (j < kSEQ) ? (_Float16)(scr[i * kSEQP + j] * inv) : (_Float16)0.f;
        } else {
          for (int j = 0; j < kKATT; ++j) att[i * kKATT + j] = (_Float16)0.f;
        }
      }
      __syncthreads();
      // O_h = att @ V -> hb[:, h*16 .. ] (9x1 tiles, 5 K-steps)
      for (int mi = wave; mi < 9; mi += 8) {
        f8v acc = {};
        for (int k0 = 0; k0 < kKATT; k0 += 32)
          acc = wmma16(ldsA(att, mi * 16, kKATT, k0), ldsBkn(Vs, 0, kHD, k0), acc);
        int nn = lane & 15, hi = lane >> 4;
#pragma unroll
        for (int v = 0; v < 8; ++v)
          hb[(mi * 16 + v + hi * 8) * kFH + h * kHD + nn] = (_Float16)acc[v];
      }
      __syncthreads();
    }

    // ---- out projection + residual ----
    for (int i = tid; i < kFH * kFH; i += 256) wb[i] = (_Float16)lp.out_w[i];
    __syncthreads();
    for (int tix = wave; tix < 9 * 4; tix += 8) {
      int mi = tix / 4, nj = tix % 4;
      f8v acc = {};
      for (int k0 = 0; k0 < kFH; k0 += 32)
        acc = wmma16(ldsA(hb, mi * 16, kFH, k0), ldsBw(wb, nj * 16, kFH, k0), acc);
      int nn = nj * 16 + (lane & 15), hi = lane >> 4;
      float bias = lp.out_b[nn];
#pragma unroll
      for (int v = 0; v < 8; ++v)
        tok[(mi * 16 + v + hi * 8) * kFH + nn] += acc[v] + bias;
    }
    __syncthreads();

    // ---- LN2 -> hb; FF1 -> relu -> ffh; FF2 + residual ----
    enc_layernorm(tok, hb, lp.ln2_g, lp.ln2_b);
    for (int i = tid; i < kFF * kFH; i += 256) wb[i] = (_Float16)lp.ff1_w[i];
    __syncthreads();
    for (int tix = wave; tix < 9 * 16; tix += 8) {
      int mi = tix / 16, nj = tix % 16;
      f8v acc = {};
      for (int k0 = 0; k0 < kFH; k0 += 32)
        acc = wmma16(ldsA(hb, mi * 16, kFH, k0), ldsBw(wb, nj * 16, kFH, k0), acc);
      int nn = nj * 16 + (lane & 15), hi = lane >> 4;
      float bias = lp.ff1_b[nn];
#pragma unroll
      for (int v = 0; v < 8; ++v)
        ffh[(mi * 16 + v + hi * 8) * kFF + nn] = (_Float16)fmaxf(acc[v] + bias, 0.f);
    }
    __syncthreads();
    for (int i = tid; i < kFH * kFF; i += 256) wb[i] = (_Float16)lp.ff2_w[i];
    __syncthreads();
    for (int tix = wave; tix < 9 * 4; tix += 8) {
      int mi = tix / 4, nj = tix % 4;
      f8v acc = {};
      for (int k0 = 0; k0 < kFF; k0 += 32)
        acc = wmma16(ldsA(ffh, mi * 16, kFF, k0), ldsBw(wb, nj * 16, kFF, k0), acc);
      int nn = nj * 16 + (lane & 15), hi = lane >> 4;
      float bias = lp.ff2_b[nn];
#pragma unroll
      for (int v = 0; v < 8; ++v)
        tok[(mi * 16 + v + hi * 8) * kFH + nn] += acc[v] + bias;
    }
    __syncthreads();
  }

  // ---- mean pool over valid tokens + output projection + relu ----
  float* pooled = scr;
  if (tid < kFH) {
    float sum = 0.f;
    for (int f = 0; f < kSEQ; ++f) sum += tok[f * kFH + tid];
    pooled[tid] = sum * (1.f / kSEQ);
  }
  __syncthreads();
  if (tid < kFH) {
    float acc = p.outp_b[tid];
    for (int c = 0; c < kFH; ++c) acc += pooled[c] * p.outp_w[tid * kFH + c];
    out[s * kFH + tid] = fmaxf(acc, 0.f);
  }
}

// ---------------- 3) frequency fuse ----------------
__global__ void freqfuse_kernel(const float* __restrict__ em, const float* __restrict__ ep,
                                const float* __restrict__ w, const float* __restrict__ b,
                                float* __restrict__ out) {
  int i = blockIdx.x * blockDim.x + threadIdx.x;
  if (i >= kS * 64) return;
  int s = i >> 6, o = i & 63;
  float acc = b[o];
  for (int c = 0; c < 64; ++c) acc += em[s * 64 + c] * w[o * 128 + c];
  for (int c = 0; c < 64; ++c) acc += ep[s * 64 + c] * w[o * 128 + 64 + c];
  out[i] = fmaxf(acc, 0.f);
}

// ---------------- 4) spatial branch ----------------
__global__ void spin_kernel(const float* __restrict__ x, const float* __restrict__ w,
                            const float* __restrict__ b, float* __restrict__ nf) {
  int i = blockIdx.x * blockDim.x + threadIdx.x;
  if (i >= kS * kGIN) return;
  int s = i / kGIN, g = i % kGIN;
  int bb = s / kN, n = s % kN;
  float acc = b[g];
  for (int t = 0; t < kT; ++t) acc += x[((size_t)bb * kT + t) * kN + n] * w[g * kT + t];
  nf[i] = acc;
}

__global__ void gatproj_kernel(const float* __restrict__ xin, int din,
                               const float* __restrict__ W, float* __restrict__ proj) {
  int i = blockIdx.x * blockDim.x + threadIdx.x;
  if (i >= kS * 256) return;
  int s = i >> 8, j = i & 255;
  float acc = 0.f;
  for (int k = 0; k < din; ++k) acc += xin[s * din + k] * W[j * din + k];
  proj[i] = acc;
}

__global__ void gatsd_kernel(const float* __restrict__ proj, const float* __restrict__ a_src,
                             const float* __restrict__ a_dst, float* __restrict__ srcb,
                             float* __restrict__ dstb) {
  int i = blockIdx.x * blockDim.x + threadIdx.x;
  if (i >= kB * kNH * kN) return;
  int n = i % kN, bh = i / kN, h = bh % kNH, bb = bh / kNH;
  const float* pr = proj + ((size_t)(bb * kN + n) * kNH + h) * 64;
  float as = 0.f, ad = 0.f;
  for (int d = 0; d < 64; ++d) { as += pr[d] * a_src[h * 64 + d]; ad += pr[d] * a_dst[h * 64 + d]; }
  srcb[i] = as;
  dstb[i] = ad;
}

__global__ void __launch_bounds__(256) gatagg_kernel(const float* __restrict__ proj,
    const float* __restrict__ srcb, const float* __restrict__ dstb,
    const int* __restrict__ adj, const float* __restrict__ bias,
    float* __restrict__ out, int do_elu) {
  const int s = blockIdx.x;  // b*N + i
  const int bb = s / kN, i = s % kN;
  __shared__ float wrow[kN];
  __shared__ float red[256];
  __shared__ float accd[64];
  const int tid = threadIdx.x;
  if (tid < 64) accd[tid] = 0.f;
  for (int h = 0; h < kNH; ++h) {
    const float si = srcb[(bb * kNH + h) * kN + i];
    const float* dsth = dstb + (bb * kNH + h) * kN;
    float lmax = -3.0e38f;
    for (int j = tid; j < kN; j += 256) {
      float e;
      if (adj[(size_t)i * kN + j]) {
        float v = si + dsth[j];
        e = (v > 0.f) ? v : 0.2f * v;  // leaky_relu(0.2)
      } else e = -3.0e38f;
      wrow[j] = e;
      lmax = fmaxf(lmax, e);
    }
    red[tid] = lmax;
    __syncthreads();
    for (int off = 128; off > 0; off >>= 1) {
      if (tid < off) red[tid] = fmaxf(red[tid], red[tid + off]);
      __syncthreads();
    }
    const float mx = red[0];
    __syncthreads();
    float lsum = 0.f;
    for (int j = tid; j < kN; j += 256) {
      float e = (wrow[j] > -1.0e37f) ? __expf(wrow[j] - mx) : 0.f;
      wrow[j] = e;
      lsum += e;
    }
    red[tid] = lsum;
    __syncthreads();
    for (int off = 128; off > 0; off >>= 1) {
      if (tid < off) red[tid] += red[tid + off];
      __syncthreads();
    }
    const float inv = 1.f / red[0];
    __syncthreads();
    if (tid < 64) {
      float acc = 0.f;
      const float* pj = proj + (size_t)bb * kN * 256 + h * 64 + tid;
      for (int j = 0; j < kN; ++j) acc += wrow[j] * pj[(size_t)j * 256];
      accd[tid] += acc * inv;
    }
    __syncthreads();
  }
  if (tid < 64) {
    float v = accd[tid] * (1.f / kNH) + bias[tid];
    if (do_elu) v = (v > 0.f) ? v : (__expf(v) - 1.f);
    out[s * 64 + tid] = v;
  }
}

// ---------------- 5) temporal branch (WMMA LSTM, 16 seqs/block) ----------------
struct LstmP { const float *w_ih, *w_hh, *b_ih, *b_hh; };

constexpr int L1_WIH  = 0;                        // f16 [256][32] (K padded 16->32)
constexpr int L1_WHH  = L1_WIH + 256 * 32 * 2;    // f16 [256][64]
constexpr int L1_BIAS = L1_WHH + 256 * 64 * 2;    // f32 [256]
constexpr int L1_XS   = L1_BIAS + 256 * 4;        // f32 [16][256]
constexpr int L1_AF   = L1_XS + 16 * 256 * 4;     // f16 [16][32]
constexpr int L1_HF   = L1_AF + 16 * 32 * 2;      // f16 [16][64]
constexpr int L1_G    = L1_HF + 16 * 64 * 2;      // f32 [16][256]
constexpr int L1_H    = L1_G + 16 * 256 * 4;      // f32 [16][64]
constexpr int L1_C    = L1_H + 16 * 64 * 4;       // f32 [16][64]
constexpr int L1_LDS  = L1_C + 16 * 64 * 4;       // 94208 B

__global__ void __launch_bounds__(256) lstm1_kernel(const float* __restrict__ x,
    const float* __restrict__ t_in_w, const float* __restrict__ t_in_b,
    LstmP pf, LstmP pb, _Float16* __restrict__ hs1) {
  extern __shared__ char smem[];
  _Float16* wih = (_Float16*)(smem + L1_WIH);
  _Float16* whh = (_Float16*)(smem + L1_WHH);
  float* bias   = (float*)(smem + L1_BIAS);
  float* xs     = (float*)(smem + L1_XS);
  _Float16* Af  = (_Float16*)(smem + L1_AF);
  _Float16* Hf  = (_Float16*)(smem + L1_HF);
  float* g      = (float*)(smem + L1_G);
  float* hsb    = (float*)(smem + L1_H);
  float* csb    = (float*)(smem + L1_C);
  const int tid = threadIdx.x, wave = tid >> 5, lane = tid & 31;
  const int grp = blockIdx.x >> 1, dir = blockIdx.x & 1;
  const int sbase = grp * 16;
  LstmP P = dir ? pb : pf;

  for (int i = tid; i < 256 * 32; i += 256) {
    int j = i >> 5, k = i & 31;
    wih[i] = (k < kLIN) ? (_Float16)P.w_ih[j * kLIN + k] : (_Float16)0.f;
  }
  for (int i = tid; i < 256 * 64; i += 256) whh[i] = (_Float16)P.w_hh[i];
  for (int i = tid; i < 256; i += 256) bias[i] = P.b_ih[i] + P.b_hh[i];
  for (int i = tid; i < 16 * kT; i += 256) {
    int sl = i >> 8, t = i & 255;
    int seq = sbase + sl, bb = seq / kN, n = seq % kN;
    xs[i] = x[((size_t)bb * kT + t) * kN + n];
  }
  for (int i = tid; i < 16 * 64; i += 256) { hsb[i] = 0.f; csb[i] = 0.f; }
  __syncthreads();

  for (int step = 0; step < kT; ++step) {
    const int t = dir ? (kT - 1 - step) : step;
    for (int i = tid; i < 16 * 32; i += 256) {
      int sl = i >> 5, l2 = i & 31;
      Af[i] = (l2 < kLIN) ? (_Float16)(xs[sl * kT + t] * t_in_w[l2] + t_in_b[l2]) : (_Float16)0.f;
    }
    for (int i = tid; i < 16 * 64; i += 256) Hf[i] = (_Float16)hsb[i];
    __syncthreads();
    // gates = X@Wih^T + H@Whh^T + b   (one M=16 tile x 16 N tiles)
    for (int nj = wave; nj < 16; nj += 8) {
      f8v acc = {};
      acc = wmma16(ldsA(Af, 0, 32, 0), ldsBw(wih, nj * 16, 32, 0), acc);
      for (int k0 = 0; k0 < 64; k0 += 32)
        acc = wmma16(ldsA(Hf, 0, 64, k0), ldsBw(whh, nj * 16, 64, k0), acc);
      int nn = nj * 16 + (lane & 15), hi = lane >> 4;
#pragma unroll
      for (int v = 0; v < 8; ++v)
        g[(v + hi * 8) * 256 + nn] = acc[v] + bias[nn];
    }
    __syncthreads();
    for (int i = tid; i < 16 * 64; i += 256) {
      int sl = i >> 6, k = i & 63;
      float gi = g[sl * 256 + k], gf = g[sl * 256 + 64 + k];
      float gg = g[sl * 256 + 128 + k], go = g[sl * 256 + 192 + k];
      float c = sigmoidf(gf) * csb[i] + sigmoidf(gi) * tanhf(gg);
      csb[i] = c;
      float h = sigmoidf(go) * tanhf(c);
      hsb[i] = h;
      hs1[((size_t)(sbase + sl) * kT + t) * 128 + dir * 64 + k] = (_Float16)h;
    }
    __syncthreads();
  }
}

constexpr int L2_WIHF = 0;                          // f16 [256][128]
constexpr int L2_WHH  = L2_WIHF + 256 * 128 * 2;    // f16 [256][64]
constexpr int L2_WIHB = L2_WHH + 256 * 64 * 2;      // f16 [256][128]
constexpr int L2_BF   = L2_WIHB + 256 * 128 * 2;    // f32 [256]
constexpr int L2_BB   = L2_BF + 1024;               // f32 [256]
constexpr int L2_AF   = L2_BB + 1024;               // f16 [16][128]
constexpr int L2_HF   = L2_AF + 16 * 128 * 2;       // f16 [16][64]
constexpr int L2_G    = L2_HF + 16 * 64 * 2;        // f32 [16][256]
constexpr int L2_H    = L2_G + 16 * 256 * 4;        // f32 [16][64]
constexpr int L2_C    = L2_H + 4096;                // f32 [16][64]
constexpr int L2_HB   = L2_C + 4096;                // f32 [16][64]
constexpr int L2_LDS  = L2_HB + 4096;               // 200704 B

__global__ void __launch_bounds__(256) lstm2_kernel(const _Float16* __restrict__ hs1,
    LstmP pf, LstmP pb, const float* __restrict__ t_out_w, const float* __restrict__ t_out_b,
    float* __restrict__ tb) {
  extern __shared__ char smem[];
  _Float16* wihf = (_Float16*)(smem + L2_WIHF);
  _Float16* whh  = (_Float16*)(smem + L2_WHH);
  _Float16* wihb = (_Float16*)(smem + L2_WIHB);
  float* bf = (float*)(smem + L2_BF);
  float* bb = (float*)(smem + L2_BB);
  _Float16* Af = (_Float16*)(smem + L2_AF);
  _Float16* Hf = (_Float16*)(smem + L2_HF);
  float* g   = (float*)(smem + L2_G);
  float* hsf = (float*)(smem + L2_H);
  float* csf = (float*)(smem + L2_C);
  float* hbk = (float*)(smem + L2_HB);
  const int tid = threadIdx.x, wave = tid >> 5, lane = tid & 31;
  const int sbase = blockIdx.x * 16;

  for (int i = tid; i < 256 * 128; i += 256) { wihf[i] = (_Float16)pf.w_ih[i]; wihb[i] = (_Float16)pb.w_ih[i]; }
  for (int i = tid; i < 256 * 64; i += 256) whh[i] = (_Float16)pf.w_hh[i];
  for (int i = tid; i < 256; i += 256) { bf[i] = pf.b_ih[i] + pf.b_hh[i]; bb[i] = pb.b_ih[i] + pb.b_hh[i]; }
  for (int i = tid; i < 16 * 64; i += 256) { hsf[i] = 0.f; csf[i] = 0.f; }
  __syncthreads();

  // forward direction: full scan, keep only final hidden
  for (int t = 0; t < kT; ++t) {
    for (int i = tid; i < 16 * 128; i += 256) {
      int sl = i >> 7, k = i & 127;
      Af[i] = hs1[((size_t)(sbase + sl) * kT + t) * 128 + k];
    }
    for (int i = tid; i < 16 * 64; i += 256) Hf[i] = (_Float16)hsf[i];
    if (tid < 16 && t + 1 < kT)
      __builtin_prefetch(&hs1[((size_t)(sbase + tid) * kT + (t + 1)) * 128], 0, 2);
    __syncthreads();
    for (int nj = wave; nj < 16; nj += 8) {
      f8v acc = {};
      for (int k0 = 0; k0 < 128; k0 += 32)
        acc = wmma16(ldsA(Af, 0, 128, k0), ldsBw(wihf, nj * 16, 128, k0), acc);
      for (int k0 = 0; k0 < 64; k0 += 32)
        acc = wmma16(ldsA(Hf, 0, 64, k0), ldsBw(whh, nj * 16, 64, k0), acc);
      int nn = nj * 16 + (lane & 15), hi = lane >> 4;
#pragma unroll
      for (int v = 0; v < 8; ++v)
        g[(v + hi * 8) * 256 + nn] = acc[v] + bf[nn];
    }
    __syncthreads();
    for (int i = tid; i < 16 * 64; i += 256) {
      int sl = i >> 6, k = i & 63;
      float gi = g[sl * 256 + k], gf = g[sl * 256 + 64 + k];
      float gg = g[sl * 256 + 128 + k], go = g[sl * 256 + 192 + k];
      float c = sigmoidf(gf) * csf[i] + sigmoidf(gi) * tanhf(gg);
      csf[i] = c;
      hsf[i] = sigmoidf(go) * tanhf(c);
    }
    __syncthreads();
  }

  // backward direction: s[:, -1] only needs ONE cell step from zero state on input t=T-1
  for (int i = tid; i < 16 * 128; i += 256) {
    int sl = i >> 7, k = i & 127;
    Af[i] = hs1[((size_t)(sbase + sl) * kT + (kT - 1)) * 128 + k];
  }
  __syncthreads();
  for (int nj = wave; nj < 16; nj += 8) {
    f8v acc = {};
    for (int k0 = 0; k0 < 128; k0 += 32)
      acc = wmma16(ldsA(Af, 0, 128, k0), ldsBw(wihb, nj * 16, 128, k0), acc);
    int nn = nj * 16 + (lane & 15), hi = lane >> 4;
#pragma unroll
    for (int v = 0; v < 8; ++v)
      g[(v + hi * 8) * 256 + nn] = acc[v] + bb[nn];
  }
  __syncthreads();
  for (int i = tid; i < 16 * 64; i += 256) {
    int sl = i >> 6, k = i & 63;
    float gi = g[sl * 256 + k];
    float gg = g[sl * 256 + 128 + k], go = g[sl * 256 + 192 + k];
    float c = sigmoidf(gi) * tanhf(gg);  // c0 = 0
    hbk[i] = sigmoidf(go) * tanhf(c);
  }
  __syncthreads();

  // t_out: relu([h_fwd, h_bwd] @ t_out_w^T + b)
  for (int i = tid; i < 16 * 64; i += 256) {
    int sl = i >> 6, o = i & 63;
    float acc = t_out_b[o];
    for (int k = 0; k < 64; ++k) acc += hsf[sl * 64 + k] * t_out_w[o * 128 + k];
    for (int k = 0; k < 64; ++k) acc += hbk[sl * 64 + k] * t_out_w[o * 128 + 64 + k];
    tb[(sbase + sl) * 64 + o] = fmaxf(acc, 0.f);
  }
}

// ---------------- 6) fusion, pooling, head ----------------
__global__ void fuse_kernel(const float* __restrict__ tb, const float* __restrict__ fb,
                            const float* __restrict__ sb, const float* __restrict__ w,
                            const float* __restrict__ bias, float* __restrict__ fused) {
  int i = blockIdx.x * blockDim.x + threadIdx.x;
  if (i >= kS * 64) return;
  int s = i >> 6, o = i & 63;
  float acc = bias[o];
  for (int c = 0; c < 64; ++c) acc += tb[s * 64 + c] * w[o * 192 + c];
  for (int c = 0; c < 64; ++c) acc += fb[s * 64 + c] * w[o * 192 + 64 + c];
  for (int c = 0; c < 64; ++c) acc += sb[s * 64 + c] * w[o * 192 + 128 + c];
  fused[i] = fmaxf(acc, 0.f);
}

__global__ void pool_kernel(const float* __restrict__ fused, float* __restrict__ pooled) {
  int bb = blockIdx.x, o = threadIdx.x;
  if (o < 64) {
    float s = 0.f;
    for (int n = 0; n < kN; ++n) s += fused[((size_t)bb * kN + n) * 64 + o];
    pooled[bb * 64 + o] = s * (1.f / kN);
  }
}

__global__ void head_kernel(const float* __restrict__ pooled, const float* __restrict__ w,
                            const float* __restrict__ b, float* __restrict__ out) {
  int i = blockIdx.x * blockDim.x + threadIdx.x;
  if (i >= kB * kN) return;
  int bb = i / kN, n = i % kN;
  float acc = b[n];
  for (int c = 0; c < 64; ++c) acc += pooled[bb * 64 + c] * w[n * 64 + c];
  out[i] = acc;  // (B, HOR=1, N) flat
}

// ---------------- host launcher ----------------
extern "C" void kernel_launch(void* const* d_in, const int* in_sizes, int n_in,
                              void* d_out, int out_size, void* d_ws, size_t ws_size,
                              hipStream_t stream) {
  (void)in_sizes; (void)n_in; (void)out_size; (void)ws_size;
  auto F = [&](int i) { return (const float*)d_in[i]; };
  const float* x = F(0);
  const int* adj = (const int*)d_in[1];

  auto mkenc = [&](int base) {
    EncP p;
    p.in_w = F(base + 0); p.in_b = F(base + 1); p.pos = F(base + 2);
    for (int l = 0; l < 2; ++l) {
      int lb = base + 3 + 12 * l;
      p.L[l].qkv_w = F(lb + 0);  p.L[l].qkv_b = F(lb + 1);
      p.L[l].out_w = F(lb + 2);  p.L[l].out_b = F(lb + 3);
      p.L[l].ln1_g = F(lb + 4);  p.L[l].ln1_b = F(lb + 5);
      p.L[l].ln2_g = F(lb + 6);  p.L[l].ln2_b = F(lb + 7);
      p.L[l].ff1_w = F(lb + 8);  p.L[l].ff1_b = F(lb + 9);
      p.L[l].ff2_w = F(lb + 10); p.L[l].ff2_b = F(lb + 11);
    }
    p.outp_w = F(base + 27); p.outp_b = F(base + 28);
    return p;
  };
  EncP magP = mkenc(2), phP = mkenc(31);
  const float* freq_fuse_w = F(60); const float* freq_fuse_b = F(61);
  const float* sp_in_w = F(62);     const float* sp_in_b = F(63);
  const float* gat0W = F(64); const float* gat0as = F(65); const float* gat0ad = F(66); const float* gat0b = F(67);
  const float* gat1W = F(68); const float* gat1as = F(69); const float* gat1ad = F(70); const float* gat1b = F(71);
  const float* t_in_w = F(72); const float* t_in_b = F(73);
  LstmP l1f = { F(74), F(75), F(76), F(77) };
  LstmP l1b = { F(78), F(79), F(80), F(81) };
  LstmP l2f = { F(82), F(83), F(84), F(85) };
  LstmP l2b = { F(86), F(87), F(88), F(89) };
  const float* t_out_w = F(90); const float* t_out_b = F(91);
  const float* fuse_w = F(92);  const float* fuse_b = F(93);
  const float* head_w = F(94);  const float* head_b = F(95);

  // workspace carve
  char* wsp = (char*)d_ws;
  size_t off = 0;
  auto carve = [&](size_t bytes) { char* p = wsp + off; off += (bytes + 255) & ~(size_t)255; return (void*)p; };
  float* magF   = (float*)carve((size_t)kS * kFB * 4);
  float* phF    = (float*)carve((size_t)kS * kFB * 4);
  float* encMag = (float*)carve((size_t)kS * 64 * 4);
  float* encPh  = (float*)carve((size_t)kS * 64 * 4);
  float* fbuf   = (float*)carve((size_t)kS * 64 * 4);
  float* nf     = (float*)carve((size_t)kS * kGIN * 4);
  float* proj   = (float*)carve((size_t)kS * 256 * 4);
  float* srcb   = (float*)carve((size_t)kB * kNH * kN * 4);
  float* dstb   = (float*)carve((size_t)kB * kNH * kN * 4);
  float* g1out  = (float*)carve((size_t)kS * 64 * 4);
  float* sbuf   = (float*)carve((size_t)kS * 64 * 4);
  float* tbuf   = (float*)carve((size_t)kS * 64 * 4);
  float* fused  = (float*)carve((size_t)kS * 64 * 4);
  float* pooled = (float*)carve((size_t)kB * 64 * 4);
  _Float16* hs1 = (_Float16*)carve((size_t)kS * kT * 128 * 2);

  // frequency branch
  dft_kernel<<<kS, 256, 0, stream>>>(x, magF, phF);
  encoder_kernel<<<kS, 256, ENC_LDS, stream>>>(magF, magP, encMag);
  encoder_kernel<<<kS, 256, ENC_LDS, stream>>>(phF, phP, encPh);
  freqfuse_kernel<<<(kS * 64 + 255) / 256, 256, 0, stream>>>(encMag, encPh, freq_fuse_w, freq_fuse_b, fbuf);

  // spatial branch
  spin_kernel<<<(kS * kGIN + 255) / 256, 256, 0, stream>>>(x, sp_in_w, sp_in_b, nf);
  gatproj_kernel<<<(kS * 256 + 255) / 256, 256, 0, stream>>>(nf, kGIN, gat0W, proj);
  gatsd_kernel<<<(kB * kNH * kN + 255) / 256, 256, 0, stream>>>(proj, gat0as, gat0ad, srcb, dstb);
  gatagg_kernel<<<kS, 256, 0, stream>>>(proj, srcb, dstb, adj, gat0b, g1out, 1);
  gatproj_kernel<<<(kS * 256 + 255) / 256, 256, 0, stream>>>(g1out, 64, gat1W, proj);
  gatsd_kernel<<<(kB * kNH * kN + 255) / 256, 256, 0, stream>>>(proj, gat1as, gat1ad, srcb, dstb);
  gatagg_kernel<<<kS, 256, 0, stream>>>(proj, srcb, dstb, adj, gat1b, sbuf, 0);

  // temporal branch (128 groups of 16 sequences; layer1 fwd/bwd as separate blocks)
  lstm1_kernel<<<256, 256, L1_LDS, stream>>>(x, t_in_w, t_in_b, l1f, l1b, hs1);
  lstm2_kernel<<<128, 256, L2_LDS, stream>>>(hs1, l2f, l2b, t_out_w, t_out_b, tbuf);

  // fuse + pool + head
  fuse_kernel<<<(kS * 64 + 255) / 256, 256, 0, stream>>>(tbuf, fbuf, sbuf, fuse_w, fuse_b, fused);
  pool_kernel<<<kB, 64, 0, stream>>>(fused, pooled);
  head_kernel<<<(kB * kN + 255) / 256, 256, 0, stream>>>(pooled, head_w, head_b, (float*)d_out);
}
